// HydraLoRAQuantLinear_39943195853506
// MI455X (gfx1250) — compile-verified, matrix operand order checked by reading, not run on hardware
//
#include <hip/hip_runtime.h>

#define IN_DIM   1024
#define OUT_DIM  1024
#define KAUG     1088          // 1024 + E*R
#define NTOK     32768         // B*S
#define EXPERTS  8
#define RANK     8
#define SCALING  2.0f

typedef __bf16 bf16;
typedef __attribute__((ext_vector_type(16))) __bf16 v16bf;
typedef __attribute__((ext_vector_type(8)))  __bf16 v8bf;
typedef __attribute__((ext_vector_type(8)))  float  v8f;

// Low 32 bits of a generic pointer into __shared__ == LDS byte offset
__device__ __forceinline__ unsigned ldsByteOffset(const void* p) {
  return (unsigned)(size_t)p;
}

// CDNA5 async copy: global -> LDS, 16B per lane, tracked by ASYNCcnt
__device__ __forceinline__ void asyncLoadB128(unsigned ldsOff, unsigned vOff,
                                              const void* sBase) {
  asm volatile("global_load_async_to_lds_b128 %0, %1, %2"
               :: "v"(ldsOff), "v"(vOff), "s"(sBase) : "memory");
}
// Async ops retire in order: waiting ASYNCcnt<=4 ensures the oldest tile
// (4 ops/thread) has fully landed while the newest 4 stay in flight.
__device__ __forceinline__ void waitAsync4() {
  asm volatile("s_wait_asynccnt 4" ::: "memory");
}
__device__ __forceinline__ void waitAsync0() {
  asm volatile("s_wait_asynccnt 0" ::: "memory");
}

// ---------------------------------------------------------------------------
// Kernel 1: pack augmented weights Wb[n][k] (bf16, N-major):
//   k < 1024        -> base_w[n][k]
//   k = 1024+e*8+r  -> lora_B[e][r][n]
// ---------------------------------------------------------------------------
__global__ __launch_bounds__(256) void hydra_pack_w(
    const float* __restrict__ base_w, const float* __restrict__ lora_B,
    bf16* __restrict__ Wb) {
  const int n = blockIdx.x;
  const int t = threadIdx.x;
  const float* src = base_w + (size_t)n * IN_DIM;
  bf16* dst = Wb + (size_t)n * KAUG;
  for (int k = t; k < IN_DIM; k += 256) dst[k] = (bf16)src[k];
  if (t < EXPERTS * RANK)
    dst[IN_DIM + t] = (bf16)lora_B[(size_t)t * OUT_DIM + n];
}

// ---------------------------------------------------------------------------
// Kernel 2: per-token prep -> augmented activation row Aaug[token][1088]
//   [0,1024)    : bf16(x)
//   [1024,1088) : bf16( softmax_bf16(x@router_w^T + rb)[e] * (x@lora_A)[r] * 2 )
// One wave per token (wave32), 8 tokens per block.
// ---------------------------------------------------------------------------
__global__ __launch_bounds__(256) void hydra_prep(
    const float* __restrict__ x, const float* __restrict__ lora_A,
    const float* __restrict__ router_w, const float* __restrict__ router_b,
    bf16* __restrict__ Aaug) {
  __shared__ __align__(16) float sA[IN_DIM][8];   // lora_A [k][r]
  __shared__ __align__(16) float sR[IN_DIM][8];   // router_w transposed [k][e]
  const int t = threadIdx.x;
  for (int i = t; i < IN_DIM * 8; i += 256) {
    sA[i >> 3][i & 7] = lora_A[i];              // lora_A is [k][r] contiguous
    sR[i & (IN_DIM - 1)][i >> 10] = router_w[i]; // router_w is [e][k]
  }
  __syncthreads();

  const int lane = t & 31, wid = t >> 5;
  const size_t token = (size_t)blockIdx.x * 8 + wid;
  const float* xr = x + token * IN_DIM;

  float lg[8] = {}, mid[8] = {};
  for (int i = 0; i < 32; ++i) {
    const int k = lane + (i << 5);       // coalesced x reads
    const float xv = xr[k];
#pragma unroll
    for (int e = 0; e < 8; ++e) lg[e]  = fmaf(xv, sR[k][e], lg[e]);
#pragma unroll
    for (int r = 0; r < 8; ++r) mid[r] = fmaf(xv, sA[k][r], mid[r]);
  }
#pragma unroll
  for (int e = 0; e < 8; ++e) {
#pragma unroll
    for (int off = 16; off > 0; off >>= 1) {
      lg[e]  += __shfl_xor(lg[e],  off, 32);
      mid[e] += __shfl_xor(mid[e], off, 32);
    }
    lg[e] += router_b[e];
  }
  float mx = lg[0];
#pragma unroll
  for (int e = 1; e < 8; ++e) mx = fmaxf(mx, lg[e]);
  float g[8], s = 0.f;
#pragma unroll
  for (int e = 0; e < 8; ++e) { g[e] = __expf(lg[e] - mx); s += g[e]; }
  const float inv = 1.0f / s;
#pragma unroll
  for (int e = 0; e < 8; ++e) g[e] = (float)(bf16)(g[e] * inv);  // bf16 gate

  bf16* arow = Aaug + token * KAUG;
  for (int i = 0; i < 32; ++i) {
    const int k = lane + (i << 5);
    arow[k] = (bf16)xr[k];
  }
#pragma unroll
  for (int u = 0; u < 2; ++u) {
    const int j = lane * 2 + u;          // j = e*8 + r, 64 values per wave
    arow[IN_DIM + j] = (bf16)(g[j >> 3] * mid[j & 7] * SCALING);
  }
}

// ---------------------------------------------------------------------------
// Kernel 3: bf16 WMMA GEMM  out[32768][1024] = Aaug[M][1088] * Wb[N][1088]^T + bias
// 128x128 tile / workgroup, 8 waves (4 M x 2 N), wave does 32x64 via 2x4 WMMAs.
// 3-stage pipeline: tile kb+2 prefetched via global_load_async_to_lds_b128
// while tile kb is consumed; s_wait_asynccnt 4 drains only the oldest tile.
// ---------------------------------------------------------------------------
__global__ __launch_bounds__(256) void hydra_gemm(
    const bf16* __restrict__ Aaug, const bf16* __restrict__ Wb,
    const float* __restrict__ bias, float* __restrict__ out) {
  // [stage][A/B][row][k + 8 pad] -> 60KB, rows 80B apart (conflict-free b128)
  __shared__ __align__(16) bf16 sm[3][2][128][40];

  const int t = threadIdx.x;
  const int lane = t & 31, wid = t >> 5;
  const int wm = wid & 3, wn = wid >> 2;
  const int bm = blockIdx.y, bn = blockIdx.x;

  const char* Ab = (const char*)(Aaug + (size_t)bm * 128 * KAUG);
  const char* Bb = (const char*)(Wb   + (size_t)bn * 128 * KAUG);

  // 512 16B-chunks per 128x32 tile -> 2 chunks/thread/matrix
  const int c0 = t, c1 = t + 256;
  const int r0 = c0 >> 2, q0 = c0 & 3;
  const int r1 = c1 >> 2, q1 = c1 & 3;
  const unsigned gv0 = (unsigned)(r0 * (KAUG * 2) + q0 * 16);
  const unsigned gv1 = (unsigned)(r1 * (KAUG * 2) + q1 * 16);

  auto issue = [&](int buf, int kb) {    // 4 async ops / thread / tile
    const void* ab = Ab + kb * 64;       // kb*32 elems * 2B
    const void* bb = Bb + kb * 64;
    asyncLoadB128(ldsByteOffset(&sm[buf][0][r0][q0 * 8]), gv0, ab);
    asyncLoadB128(ldsByteOffset(&sm[buf][0][r1][q1 * 8]), gv1, ab);
    asyncLoadB128(ldsByteOffset(&sm[buf][1][r0][q0 * 8]), gv0, bb);
    asyncLoadB128(ldsByteOffset(&sm[buf][1][r1][q1 * 8]), gv1, bb);
  };
  // 16-bit A/B fragment: lanes 0-15 hold K{ko..+7, ko+16..+23}, lanes 16-31
  // the complementary halves (ko = 8 for upper lanes).
  auto ldfrag = [](const bf16* rowp) -> v16bf {
    v8bf lo = *(const v8bf*)(rowp);
    v8bf hi = *(const v8bf*)(rowp + 16);
    return __builtin_shufflevector(lo, hi, 0, 1, 2, 3, 4, 5, 6, 7,
                                   8, 9, 10, 11, 12, 13, 14, 15);
  };

  v8f acc[2][4] = {};
  const int lr = lane & 15;
  const int ko = (lane >> 4) * 8;

  const int KSTEPS = KAUG / 32;  // 34
  issue(0, 0);
  issue(1, 1);
  waitAsync4();                  // tile 0 resident, tile 1 may be in flight
  __syncthreads();

  int buf = 0;
  for (int kb = 0; kb < KSTEPS; ++kb) {
    if (kb + 2 < KSTEPS) {
      int nxt = buf + 2; if (nxt >= 3) nxt -= 3;
      issue(nxt, kb + 2);
    }

    const bf16(*As)[40] = sm[buf][0];
    const bf16(*Bs)[40] = sm[buf][1];
    v16bf a[2], b[4];
#pragma unroll
    for (int mf = 0; mf < 2; ++mf)
      a[mf] = ldfrag(&As[wm * 32 + mf * 16 + lr][ko]);
#pragma unroll
    for (int nf = 0; nf < 4; ++nf)
      b[nf] = ldfrag(&Bs[wn * 64 + nf * 16 + lr][ko]);
#pragma unroll
    for (int mf = 0; mf < 2; ++mf)
#pragma unroll
      for (int nf = 0; nf < 4; ++nf)
        acc[mf][nf] = __builtin_amdgcn_wmma_f32_16x16x32_bf16(
            false, a[mf], false, b[nf], (short)0, acc[mf][nf], false, false);

    // Guarantee tile kb+1 is fully in LDS before the barrier releases it.
    if (kb + 2 < KSTEPS) waitAsync4();   // keep tile kb+2's 4 ops in flight
    else                 waitAsync0();   // pipeline drain at the tail
    __syncthreads();

    if (++buf == 3) buf = 0;
  }

  // C layout: VGPR i -> row = i + 8*(lane>>4), col = lane&15
  const int colSel = lane & 15;
  const int rowSel = (lane >> 4) * 8;
#pragma unroll
  for (int nf = 0; nf < 4; ++nf) {
    const int gn = bn * 128 + wn * 64 + nf * 16 + colSel;
    const float bv = bias[gn];
#pragma unroll
    for (int mf = 0; mf < 2; ++mf) {
      const size_t gm0 = (size_t)(bm * 128 + wm * 32 + mf * 16 + rowSel);
#pragma unroll
      for (int i = 0; i < 8; ++i)
        out[(gm0 + i) * OUT_DIM + gn] = acc[mf][nf][i] + bv;
    }
  }
}

// ---------------------------------------------------------------------------
extern "C" void kernel_launch(void* const* d_in, const int* in_sizes, int n_in,
                              void* d_out, int out_size, void* d_ws,
                              size_t ws_size, hipStream_t stream) {
  (void)in_sizes; (void)n_in; (void)out_size; (void)ws_size;
  const float* x        = (const float*)d_in[0];
  const float* base_w   = (const float*)d_in[1];
  const float* base_b   = (const float*)d_in[2];
  const float* lora_A   = (const float*)d_in[3];
  const float* lora_B   = (const float*)d_in[4];
  const float* router_w = (const float*)d_in[5];
  const float* router_b = (const float*)d_in[6];
  float* out = (float*)d_out;

  bf16* Wb   = (bf16*)d_ws;                                        // 2.2 MB
  bf16* Aaug = (bf16*)((char*)d_ws + (size_t)OUT_DIM * KAUG * 2);  // 71.3 MB

  hydra_pack_w<<<OUT_DIM, 256, 0, stream>>>(base_w, lora_B, Wb);
  hydra_prep<<<NTOK / 8, 256, 0, stream>>>(x, lora_A, router_w, router_b, Aaug);
  hydra_gemm<<<dim3(OUT_DIM / 128, NTOK / 128), 256, 0, stream>>>(Aaug, Wb,
                                                                  base_b, out);
}
